// hypercorre_topk2_82008105550158
// MI455X (gfx1250) — compile-verified, matrix operand order checked by reading, not run on hardware
//
#include <hip/hip_runtime.h>

#define K_TOP  5
#define BATCH  2
#define TCLIPS 3
#define NSAMP  (BATCH * TCLIPS * K_TOP)   // 30 pyramid samples

typedef __attribute__((ext_vector_type(16))) __bf16   v16bf;
typedef __attribute__((ext_vector_type(8)))  float    v8f;
typedef __attribute__((ext_vector_type(4)))  unsigned v4u;
typedef __attribute__((ext_vector_type(8)))  int      v8i;
typedef __attribute__((ext_vector_type(4)))  int      v4i;

__device__ __forceinline__ unsigned bf_bits(float f) {   // f32 -> bf16 bits, RNE
  union { float f; unsigned u; } v; v.f = f;
  return (v.u + 0x7FFFu + ((v.u >> 16) & 1u)) >> 16;
}

// ---------------------------------------------------------------------------
// TDM: issue a 2D tensor_load_to_lds. D# per CDNA5 ISA ch.8:
//  group0: [1:0]=count=1, [63:32]=lds byte addr, [120:64]=global addr, [127:126]=type=2
//  group1: [17:16]=data_size=1(2B), [79:48]=tensor_dim0, [111:80]=tensor_dim1,
//          [127:112]=tile_dim0, [143:128]=tile_dim1, [207:160]=tensor_dim0_stride
// ---------------------------------------------------------------------------
__device__ __forceinline__ void tdm_load_2d(unsigned lds_off, const void* gptr,
                                            unsigned tile_k, unsigned tile_rows,
                                            unsigned stride_elems, unsigned td0,
                                            unsigned td1) {
  unsigned long long ga = (unsigned long long)(size_t)gptr;
  v4u g0;
  g0[0] = 1u;                                   // count=1, user descriptor
  g0[1] = lds_off;                              // LDS byte address
  g0[2] = (unsigned)(ga & 0xFFFFFFFFu);
  g0[3] = (unsigned)(ga >> 32) | (2u << 30);    // type=2 ("image")
  v8i g1;
  g1[0] = (int)(1u << 16);                      // data_size = 2 bytes
  g1[1] = (int)((td0 & 0xFFFFu) << 16);         // tensor_dim0[15:0] @ bits 63:48
  g1[2] = (int)(((td0 >> 16) & 0xFFFFu) | ((td1 & 0xFFFFu) << 16));
  g1[3] = (int)(((td1 >> 16) & 0xFFFFu) | (tile_k << 16));   // tile_dim0
  g1[4] = (int)(tile_rows & 0xFFFFu);           // tile_dim1 (tile_dim2 = 0)
  g1[5] = (int)stride_elems;                    // tensor_dim0_stride low 32
  g1[6] = 0; g1[7] = 0;
  v4i gz = {0, 0, 0, 0};
#if defined(__clang_major__) && (__clang_major__ >= 23)
  v8i gz8 = {0, 0, 0, 0, 0, 0, 0, 0};
  __builtin_amdgcn_tensor_load_to_lds(g0, g1, gz, gz, gz8, 0);
#else
  __builtin_amdgcn_tensor_load_to_lds(g0, g1, gz, gz, 0);
#endif
}

// ---------------------------------------------------------------------------
// Pre-conversion: x [chunk, c, hw] f32 -> xbf [chunk*hw, c] bf16 (transpose+cvt)
// Pair-packed u32 stores (coalesced along K).
// ---------------------------------------------------------------------------
__global__ void cvt_x_k(const float* __restrict__ x, unsigned* __restrict__ xbf,
                        int c, int hw, int total_pairs) {
  int gid = blockIdx.x * blockDim.x + threadIdx.x;
  if (gid >= total_pairs) return;
  const int cpairs = c >> 1;
  const int row = gid / cpairs, kp = gid - row * cpairs;
  const int chunk = row / hw, pix = row - chunk * hw;
  const float* xb = x + (size_t)chunk * c * hw + pix;
  unsigned lo = bf_bits(xb[(size_t)(2 * kp) * hw]);
  unsigned hi = bf_bits(xb[(size_t)(2 * kp + 1) * hw]);
  xbf[gid] = lo | (hi << 16);
}

// w [c*c] f32 -> bf16, pair-packed
__global__ void cvt_w_k(const float* __restrict__ w, unsigned* __restrict__ wbf,
                        int total_pairs) {
  int gid = blockIdx.x * blockDim.x + threadIdx.x;
  if (gid >= total_pairs) return;
  unsigned lo = bf_bits(w[2 * gid]);
  unsigned hi = bf_bits(w[2 * gid + 1]);
  wbf[gid] = lo | (hi << 16);
}

// ---------------------------------------------------------------------------
// Projection GEMM: out[row,o] = sum_k xbf[row,k]*wbf[o,k] + bias[o]
// Block = 4 waves; each block computes 64(M) x 16(N); K stepped by 32.
// TDM stages A (64x32 bf16 @ LDS 0) and B (16x32 bf16 @ LDS 4096); fragments are
// two ds_load_b128 per operand (row-major [row][32] bf16 matches the ISA 7.12.2
// 16-bit fragment map: elem e -> K = 16*(e/8) + 8*half + e%8).
// C/D map: elem r -> M = 8*half + r, N = lane&15.
// ---------------------------------------------------------------------------
__global__ void proj_gemm_wmma(const unsigned short* __restrict__ xbf,
                               const unsigned short* __restrict__ wbf,
                               const float* __restrict__ bias, float* __restrict__ out,
                               int c, int M) {
  extern __shared__ unsigned lds[];             // A: u32[0..1023], B: u32[1024..1279]
  const int lane = threadIdx.x;
  const int wy = threadIdx.y;
  const int half = lane >> 4, l = lane & 15;
  const int row0 = blockIdx.x * 64;
  const int n0 = blockIdx.y * 16;
  v8f acc = {};
  for (int kk = 0; kk < c; kk += 32) {
    __syncthreads();                            // prior iteration's fragments consumed
    if (wy == 0) {                              // wave-uniform: wave 0 drives the TDM
      tdm_load_2d(0u,    (const void*)(xbf + (size_t)row0 * c + kk), 32u, 64u,
                  (unsigned)c, (unsigned)c, (unsigned)M);
      tdm_load_2d(4096u, (const void*)(wbf + (size_t)n0 * c + kk), 32u, 16u,
                  (unsigned)c, (unsigned)c, (unsigned)c);
      __builtin_amdgcn_s_wait_tensorcnt(0);
    }
    __syncthreads();
    const int ra = (wy * 16 + l) * 16 + 4 * half;       // u32 index, row stride 16 u32
    const int rb = 1024 + l * 16 + 4 * half;
    union Frag { uint4 q[2]; v16bf v; } A, B;
    A.q[0] = *reinterpret_cast<const uint4*>(&lds[ra]);
    A.q[1] = *reinterpret_cast<const uint4*>(&lds[ra + 8]);
    B.q[0] = *reinterpret_cast<const uint4*>(&lds[rb]);
    B.q[1] = *reinterpret_cast<const uint4*>(&lds[rb + 8]);
    acc = __builtin_amdgcn_wmma_f32_16x16x32_bf16(false, A.v, false, B.v,
                                                  (short)0, acc, false, false);
  }
  const int m0 = row0 + wy * 16;
#pragma unroll
  for (int r = 0; r < 8; ++r) {
    const int Mi = m0 + (half << 3) + r;
    const int N = n0 + l;
    out[(size_t)Mi * c + N] = acc[r] + bias[N];
  }
}

// In-place L2 normalize each row of x [rows, c] (F.normalize eps=1e-12)
__global__ void l2norm_rows(float* __restrict__ x, int c) {
  __shared__ float sm[128];
  float* p = x + (size_t)blockIdx.x * c;
  float s = 0.f;
  for (int i = threadIdx.x; i < c; i += 128) { float v = p[i]; s += v * v; }
  sm[threadIdx.x] = s; __syncthreads();
  for (int st = 64; st > 0; st >>= 1) {
    if (threadIdx.x < st) sm[threadIdx.x] += sm[threadIdx.x + st];
    __syncthreads();
  }
  const float inv = 1.f / fmaxf(sqrtf(sm[0]), 1e-12f);
  for (int i = threadIdx.x; i < c; i += 128) p[i] *= inv;
}

// qmean[b,k] = mean over rows_per_b rows of qn[b, :, k];  qn: [B*rows_per_b, c]
__global__ void col_mean_k(const float* __restrict__ qn, float* __restrict__ qmean,
                           int c, int rows_per_b, int total) {
  int idx = blockIdx.x * blockDim.x + threadIdx.x;
  if (idx >= total) return;
  int b = idx / c, k = idx - b * c;
  const float* p = qn + (size_t)b * rows_per_b * c + k;
  float s = 0.f;
  for (int r = 0; r < rows_per_b; ++r) s += p[(size_t)r * c];
  qmean[idx] = s / (float)rows_per_b;
}

// score[b,n] = sn[b*hw+n,:] . qmean[b,:]   (mean-factorized correlation score)
__global__ void score_k(const float* __restrict__ sn, const float* __restrict__ qmean,
                        float* __restrict__ score, int c, int hw, int total) {
  int idx = blockIdx.x * blockDim.x + threadIdx.x;
  if (idx >= total) return;
  int b = idx / hw;
  const float* a = sn + (size_t)idx * c;
  const float* q = qmean + (size_t)b * c;
  float s = 0.f;
  for (int k = 0; k < c; ++k) s += a[k] * q[k];
  score[idx] = s;
}

// top-5 argmax per batch (destroys score); tie -> lowest index (lax.top_k order)
__global__ void top5_k(float* __restrict__ score, int* __restrict__ idx, int hw) {
  __shared__ float bv[256]; __shared__ int bi[256];
  const int b = blockIdx.x;
  float* s = score + (size_t)b * hw;
  for (int k = 0; k < K_TOP; ++k) {
    float best = -3.0e38f; int besti = hw;
    for (int i = threadIdx.x; i < hw; i += 256) {
      float v = s[i];
      if (v > best || (v == best && i < besti)) { best = v; besti = i; }
    }
    bv[threadIdx.x] = best; bi[threadIdx.x] = besti; __syncthreads();
    for (int st = 128; st > 0; st >>= 1) {
      if (threadIdx.x < st) {
        if (bv[threadIdx.x + st] > bv[threadIdx.x] ||
            (bv[threadIdx.x + st] == bv[threadIdx.x] && bi[threadIdx.x + st] < bi[threadIdx.x])) {
          bv[threadIdx.x] = bv[threadIdx.x + st]; bi[threadIdx.x] = bi[threadIdx.x + st];
        }
      }
      __syncthreads();
    }
    if (threadIdx.x == 0) { idx[b * K_TOP + k] = bi[0]; s[bi[0]] = -3.3e38f; }
    __syncthreads();
  }
}

// pyr[((b*T+t)*K_TOP+k)*hw + m] = sn[b*hw+idx[b,k],:] . qn[(b*T+t)*hw+m,:]
__global__ void sel_k(const float* __restrict__ sn, const float* __restrict__ qn,
                      const int* __restrict__ idx, float* __restrict__ pyr,
                      int c, int hw, int total) {
  int gid = blockIdx.x * blockDim.x + threadIdx.x;
  if (gid >= total) return;
  int m = gid % hw, samp = gid / hw;
  int k = samp % K_TOP, bt = samp / K_TOP;
  int tt = bt % TCLIPS, b = bt / TCLIPS;
  int n = idx[b * K_TOP + k];
  const float* a = sn + (size_t)(b * hw + n) * c;
  const float* q = qn + (size_t)((b * TCLIPS + tt) * hw + m) * c;
  float s = 0.f;
  for (int i = 0; i < c; ++i) s += a[i] * q[i];
  pyr[gid] = s;
}

// Generic small conv2d, stride 1, SAME padding (pad = k/2); NCHW / OIHW
__global__ void conv2d_k(const float* __restrict__ x, const float* __restrict__ w,
                         const float* __restrict__ bias, float* __restrict__ y,
                         int N, int Cin, int Cout, int H, int W, int ksz, int total) {
  int gid = blockIdx.x * blockDim.x + threadIdx.x;
  if (gid >= total) return;
  int xw = gid % W; int t = gid / W;
  int yy = t % H;   t /= H;
  int co = t % Cout; int n = t / Cout;
  const int pad = ksz >> 1;
  float acc = bias[co];
  for (int ci = 0; ci < Cin; ++ci) {
    const float* xb = x + ((size_t)(n * Cin + ci) * H) * W;
    const float* wb = w + ((size_t)(co * Cin + ci) * ksz) * ksz;
    for (int ky = 0; ky < ksz; ++ky) {
      int iy = yy + ky - pad;
      if (iy < 0 || iy >= H) continue;
      for (int kx = 0; kx < ksz; ++kx) {
        int ix = xw + kx - pad;
        if (ix < 0 || ix >= W) continue;
        acc += xb[(size_t)iy * W + ix] * wb[ky * ksz + kx];
      }
    }
  }
  y[gid] = acc;
}

// GroupNorm(1 group, eps=1e-5) over (C,H,W) per sample + affine + ReLU, in place
__global__ void gn_relu_k(float* __restrict__ x, const float* __restrict__ gw,
                          const float* __restrict__ gb, int C, int HW) {
  __shared__ float sm[256], sm2[256];
  const int total = C * HW;
  float* p = x + (size_t)blockIdx.x * total;
  float s = 0.f, s2 = 0.f;
  for (int i = threadIdx.x; i < total; i += 256) { float v = p[i]; s += v; s2 += v * v; }
  sm[threadIdx.x] = s; sm2[threadIdx.x] = s2; __syncthreads();
  for (int st = 128; st > 0; st >>= 1) {
    if (threadIdx.x < st) { sm[threadIdx.x] += sm[threadIdx.x + st]; sm2[threadIdx.x] += sm2[threadIdx.x + st]; }
    __syncthreads();
  }
  const float mean = sm[0] / (float)total;
  const float var  = fmaxf(sm2[0] / (float)total - mean * mean, 0.f);
  const float inv  = rsqrtf(var + 1e-5f);
  for (int i = threadIdx.x; i < total; i += 256) {
    int c = i / HW;
    float v = (p[i] - mean) * inv * gw[c] + gb[c];
    p[i] = fmaxf(v, 0.f);
  }
}

// y = bilinear_align_corners(x, Hi,Wi -> Ho,Wo) + skip
__global__ void upsample_add_k(const float* __restrict__ x, const float* __restrict__ skip,
                               float* __restrict__ y, int N, int C, int Hi, int Wi,
                               int Ho, int Wo, int total) {
  int gid = blockIdx.x * blockDim.x + threadIdx.x;
  if (gid >= total) return;
  int xo = gid % Wo; int t = gid / Wo;
  int yo = t % Ho;   t /= Ho;
  int c = t % C;     int n = t / C;
  float fy = (Ho == 1) ? 0.f : yo * (float)(Hi - 1) / (float)(Ho - 1);
  float fx = (Wo == 1) ? 0.f : xo * (float)(Wi - 1) / (float)(Wo - 1);
  int y0 = (int)floorf(fy); int y1 = min(y0 + 1, Hi - 1); float wy = fy - (float)y0;
  int x0 = (int)floorf(fx); int x1 = min(x0 + 1, Wi - 1); float wx = fx - (float)x0;
  const float* b = x + ((size_t)(n * C + c) * Hi) * Wi;
  float v = (b[y0 * Wi + x0] * (1.f - wy) + b[y1 * Wi + x0] * wy) * (1.f - wx)
          + (b[y0 * Wi + x1] * (1.f - wy) + b[y1 * Wi + x1] * wy) * wx;
  y[gid] = v + skip[gid];
}

// ---------------------------------------------------------------------------
extern "C" void kernel_launch(void* const* d_in, const int* in_sizes, int n_in,
                              void* d_out, int out_size, void* d_ws, size_t ws_size,
                              hipStream_t stream) {
  (void)out_size; (void)ws_size;
  // ---- Parameter leaf resolution (52 leaves, insertion order reference) ----
  // slots: 0..11 = q1w,q1b,k1w,k1b,q2w,q2b,k2w,k2b,q3w,q3b,k3w,k3b
  //        12..19 enc4, 20..27 enc3, 28..35 enc2, 36..43 enc43, 44..51 enc32
  static const int leaf_sz[52] = {
    16384,128,16384,128,102400,320,102400,320,262144,512,262144,512,
    9,1,1,1, 18,2,2,2,
    9,1,1,1, 18,2,2,2,
    25,1,1,1, 18,2,2,2,
    36,2,2,2, 36,2,2,2,
    36,2,2,2, 18,1,1,1
  };
  const float* L[52];
  if (n_in >= 60 && in_sizes[8] == 16384) {
    for (int i = 0; i < 52; ++i) L[i] = (const float*)d_in[8 + i];
  } else if (n_in >= 60) {
    // jax sorted-pytree order: hpn{enc2,enc3,enc32,enc4,enc43}, k1,k2,k3,q1,q2,q3
    static const int map_sorted[52] = {
      28,29,30,31,32,33,34,35,   // enc2
      20,21,22,23,24,25,26,27,   // enc3
      44,45,46,47,48,49,50,51,   // enc32
      12,13,14,15,16,17,18,19,   // enc4
      36,37,38,39,40,41,42,43,   // enc43
      2,3, 6,7, 10,11, 0,1, 4,5, 8,9
    };
    for (int i = 0; i < 52; ++i) L[map_sorted[i]] = (const float*)d_in[8 + i];
  } else {
    const float* base = (const float*)d_in[8];  // single concatenated buffer
    size_t off = 0;
    for (int i = 0; i < 52; ++i) { L[i] = base + off; off += leaf_sz[i]; }
  }

  // ---- Workspace layout (floats) ----
  float* ws = (float*)d_ws;
  size_t o = 0;
  float* qn    = ws + o; o += 3145728;   // max B*T*hw*c
  float* sn    = ws + o; o += 1048576;   // max B*hw*c
  float* qmean = ws + o; o += 1024;
  float* score = ws + o; o += 8192;
  int*   tidx  = (int*)(ws + o); o += 16;
  float* pyr0  = ws + o; o += (size_t)NSAMP * 16 * 16;
  float* pyr1  = ws + o; o += (size_t)NSAMP * 32 * 32;
  float* pyr2  = ws + o; o += (size_t)NSAMP * 64 * 64;
  float* t16a  = ws + o; o += (size_t)NSAMP * 2 * 16 * 16;
  float* t16b  = ws + o; o += (size_t)NSAMP * 2 * 16 * 16;
  float* t32a  = ws + o; o += (size_t)NSAMP * 2 * 32 * 32;
  float* t32b  = ws + o; o += (size_t)NSAMP * 2 * 32 * 32;
  float* t32c  = ws + o; o += (size_t)NSAMP * 2 * 32 * 32;
  float* t64a  = ws + o; o += (size_t)NSAMP * 2 * 64 * 64;
  float* t64b  = ws + o; o += (size_t)NSAMP * 2 * 64 * 64;
  float* t64c  = ws + o; o += (size_t)NSAMP * 2 * 64 * 64;
  unsigned short* xq_bf = (unsigned short*)(ws + o); o += 1572864;  // 24576*128 bf16
  unsigned short* xs_bf = (unsigned short*)(ws + o); o += 524288;   // 8192*128  bf16
  unsigned short* wsb   = (unsigned short*)(ws + o); o += 131072;   // 512*512   bf16
  unsigned short* wkb   = (unsigned short*)(ws + o); o += 131072;

  // ---- Per-scale correlation + top-5 selection ----
  struct Sc { const float* q; const float* s; int c; int h;
              const float* sw; const float* sb; const float* kw; const float* kb; float* pyr; };
  const Sc sc[3] = {
    { (const float*)d_in[6], (const float*)d_in[7], 512, 16, L[8], L[9], L[10], L[11], pyr0 },
    { (const float*)d_in[4], (const float*)d_in[5], 320, 32, L[4], L[5], L[6],  L[7],  pyr1 },
    { (const float*)d_in[2], (const float*)d_in[3], 128, 64, L[0], L[1], L[2],  L[3],  pyr2 },
  };
  for (int ii = 0; ii < 3; ++ii) {
    const int c = sc[ii].c, h = sc[ii].h, hw = h * h;
    const int Ms = BATCH * hw, Mq = BATCH * TCLIPS * hw;
    // bf16 pre-conversion (transpose CHW->HWC for activations)
    int tp = Ms * (c >> 1);
    cvt_x_k<<<(tp + 255) / 256, 256, 0, stream>>>(sc[ii].s, (unsigned*)xs_bf, c, hw, tp);
    tp = Mq * (c >> 1);
    cvt_x_k<<<(tp + 255) / 256, 256, 0, stream>>>(sc[ii].q, (unsigned*)xq_bf, c, hw, tp);
    tp = (c * c) >> 1;
    cvt_w_k<<<(tp + 255) / 256, 256, 0, stream>>>(sc[ii].sw, (unsigned*)wsb, tp);
    cvt_w_k<<<(tp + 255) / 256, 256, 0, stream>>>(sc[ii].kw, (unsigned*)wkb, tp);
    // WMMA projection GEMMs (TDM-staged tiles)
    dim3 blk(32, 4);
    proj_gemm_wmma<<<dim3(Ms / 64, c / 16), blk, 5120, stream>>>(xs_bf, wsb, sc[ii].sb, sn, c, Ms);
    proj_gemm_wmma<<<dim3(Mq / 64, c / 16), blk, 5120, stream>>>(xq_bf, wkb, sc[ii].kb, qn, c, Mq);
    l2norm_rows<<<Ms, 128, 0, stream>>>(sn, c);
    l2norm_rows<<<Mq, 128, 0, stream>>>(qn, c);
    int tot = BATCH * c;
    col_mean_k<<<(tot + 255) / 256, 256, 0, stream>>>(qn, qmean, c, TCLIPS * hw, tot);
    tot = BATCH * hw;
    score_k<<<(tot + 255) / 256, 256, 0, stream>>>(sn, qmean, score, c, hw, tot);
    top5_k<<<BATCH, 256, 0, stream>>>(score, tidx, hw);
    tot = BATCH * TCLIPS * K_TOP * hw;
    sel_k<<<(tot + 255) / 256, 256, 0, stream>>>(sn, qn, tidx, sc[ii].pyr, c, hw, tot);
  }

  // ---- Hypercorrelation pyramid network ----
  auto enc_block = [&](const float* in, int Cin, int Cout, int H, int W, int ksz,
                       const float* cw, const float* cbias, const float* gw, const float* gb,
                       float* out) {
    int total = NSAMP * Cout * H * W;
    conv2d_k<<<(total + 255) / 256, 256, 0, stream>>>(in, cw, cbias, out, NSAMP, Cin, Cout, H, W, ksz, total);
    gn_relu_k<<<NSAMP, 256, 0, stream>>>(out, gw, gb, Cout, H * W);
  };

  // enc4: pyr0 [30,1,16,16] -> s4 (t16b [30,2,16,16])
  enc_block(pyr0, 1, 1, 16, 16, 3, L[12], L[13], L[14], L[15], t16a);
  enc_block(t16a, 1, 2, 16, 16, 3, L[16], L[17], L[18], L[19], t16b);
  // enc3: pyr1 -> s3 (t32b [30,2,32,32])
  enc_block(pyr1, 1, 1, 32, 32, 3, L[20], L[21], L[22], L[23], t32a);
  enc_block(t32a, 1, 2, 32, 32, 3, L[24], L[25], L[26], L[27], t32b);
  // enc2: pyr2 -> s2 (t64b [30,2,64,64])
  enc_block(pyr2, 1, 1, 64, 64, 5, L[28], L[29], L[30], L[31], t64a);
  enc_block(t64a, 1, 2, 64, 64, 3, L[32], L[33], L[34], L[35], t64b);

  // upsample s4 16->32 (+ s3) -> t32c
  {
    int total = NSAMP * 2 * 32 * 32;
    upsample_add_k<<<(total + 255) / 256, 256, 0, stream>>>(t16b, t32b, t32c, NSAMP, 2, 16, 16, 32, 32, total);
  }
  // enc43: t32c -> t32a -> t32c   (m43 in t32c)
  enc_block(t32c, 2, 2, 32, 32, 3, L[36], L[37], L[38], L[39], t32a);
  enc_block(t32a, 2, 2, 32, 32, 3, L[40], L[41], L[42], L[43], t32c);

  // upsample m43 32->64 (+ s2) -> t64c
  {
    int total = NSAMP * 2 * 64 * 64;
    upsample_add_k<<<(total + 255) / 256, 256, 0, stream>>>(t32c, t64b, t64c, NSAMP, 2, 32, 32, 64, 64, total);
  }
  // enc32: t64c -> t64a -> d_out [30,1,64,64]
  enc_block(t64c, 2, 2, 64, 64, 3, L[44], L[45], L[46], L[47], t64a);
  enc_block(t64a, 2, 1, 64, 64, 3, L[48], L[49], L[50], L[51], (float*)d_out);
}
